// IRS_38714835206339
// MI455X (gfx1250) — matrix operand order repaired and checked by reading, not compiled, not run on hardware
//
#include <hip/hip_runtime.h>

#define DIMC 64
#define NB 8
#define HH 256
#define WW 256
#define HWP (HH*WW)
#define LDK 264   // padded LDS row (halves); 528 B = multiple of 16

typedef __attribute__((ext_vector_type(16))) _Float16 v16h;
typedef __attribute__((ext_vector_type(8)))  _Float16 v8h;
typedef __attribute__((ext_vector_type(8)))  float    v8f;

union Frag16 { v16h v; v8h h[2]; };

// A fragment: 16(M) x 32(K) f16, source row-major (rows = M), leading dim ld in halves.
// lanes 0-15: M=lane, K 0-7 in V0-3, K 16-23 in V4-7; lanes 16-31: K 8-15 / 24-31.
__device__ inline v16h load_a(const _Float16* p, int ld) {
  int lane = threadIdx.x & 31;
  int hf = lane >> 4;
  int m  = lane & 15;
  const _Float16* row = p + m * ld;
  Frag16 f;
  f.h[0] = *(const v8h*)(row + hf * 8);
  f.h[1] = *(const v8h*)(row + 16 + hf * 8);
  return f.v;
}

// B fragment: 32(K) x 16(N) f16, source column-major (row n contiguous over K), ld in halves.
// lanes 0-15: N=lane, K 0-7 / 16-23; lanes 16-31: N=lane-16, K 8-15 / 24-31.
__device__ inline v16h load_b(const _Float16* p, int ld) {
  int lane = threadIdx.x & 31;
  int hf = lane >> 4;
  int n  = lane & 15;
  const _Float16* row = p + n * ld;
  Frag16 f;
  f.h[0] = *(const v8h*)(row + hf * 8);
  f.h[1] = *(const v8h*)(row + 16 + hf * 8);
  return f.v;
}

#define WMMA_F16(A, Bf, C) \
  __builtin_amdgcn_wmma_f32_16x16x32_f16(false, (A), false, (Bf), (short)0, (C), false, false)

// Async-copy one 16x256-half tile (row ld 256) into padded LDS (row ld LDK).
// 512 x 16B chunks, 128 threads -> 4 async b128 copies per thread (ASYNCcnt += 4).
__device__ inline void stage_tile16(const _Float16* gsrc, _Float16* lbuf, int tid) {
#pragma unroll
  for (int i = 0; i < 4; ++i) {
    int chunk = tid + i * 128;           // 0..511
    int row   = chunk >> 5;              // 32 chunks of 16B per row
    int col   = (chunk & 31) << 3;       // halves
    unsigned   loff = (unsigned)(uintptr_t)(lbuf + row * LDK + col);
    const void* g   = (const void*)(gsrc + row * 256 + col);
    asm volatile("global_load_async_to_lds_b128 %0, %1, off"
                 :: "v"(loff), "v"(g) : "memory");
  }
}
#define WAIT_ASYNC_LE4() asm volatile("s_wait_asynccnt 0x4" ::: "memory")
#define WAIT_ASYNC_0()   asm volatile("s_wait_asynccnt 0x0" ::: "memory")

// ---------------- prep: cast weights to f16 ----------------
__global__ __launch_bounds__(256) void cvt_weights(const float* qkv_w, const float* proj_w,
                                                   _Float16* wq, _Float16* wp) {
  int i = blockIdx.x * 256 + threadIdx.x;
  if (i < 192 * 64) wq[i] = (_Float16)qkv_w[i];
  if (i < 64 * 64)  wp[i] = (_Float16)proj_w[i];
}

// ---------------- kernel 1: qkv = W[192,64] x X[64,HW] + b (per batch) ----------------
__global__ __launch_bounds__(256) void qkv_pointwise(const float* __restrict__ x,
                                                     const _Float16* __restrict__ wq,
                                                     const float* __restrict__ qkv_b,
                                                     _Float16* __restrict__ qkv1) {
  __shared__ __align__(16) _Float16 xt[128 * 72];   // x tile transposed: [p][c], pad 72
  int b    = blockIdx.x >> 9;
  int pblk = (blockIdx.x & 511) << 7;
  const float* xb = x + (size_t)b * DIMC * HWP + pblk;
  for (int idx = threadIdx.x; idx < 64 * 128; idx += 256) {
    int c = idx >> 7, p = idx & 127;
    xt[p * 72 + c] = (_Float16)xb[(size_t)c * HWP + p];
  }
  __syncthreads();
  int wave = threadIdx.x >> 5;
  int lane = threadIdx.x & 31;
  int hf = lane >> 4, ln = lane & 15;
  int pt = wave;
#pragma unroll 2   // two independent WMMA chains in flight to cover the co-exec hazard
  for (int ot = 0; ot < 12; ++ot) {
    v8f acc;
#pragma unroll
    for (int r = 0; r < 8; ++r) acc[r] = qkv_b[ot * 16 + hf * 8 + r];
#pragma unroll
    for (int kk = 0; kk < 2; ++kk) {
      v16h a  = load_a(wq + ot * 16 * 64 + kk * 32, 64);
      v16h bf = load_b(xt + pt * 16 * 72 + kk * 32, 72);
      acc = WMMA_F16(a, bf, acc);
    }
    int p = pblk + pt * 16 + ln;
#pragma unroll
    for (int r = 0; r < 8; ++r) {
      int o = ot * 16 + hf * 8 + r;
      qkv1[((size_t)b * 192 + o) * HWP + p] = (_Float16)acc[r];
    }
  }
}

// ---------------- kernel 2: 3x3 depthwise + split + L2-normalize(q,k) + transpose(v) ----
__global__ __launch_bounds__(256) void dw_norm(const _Float16* __restrict__ qkv1,
                                               const float* __restrict__ dw_w,
                                               const float* __restrict__ dw_b,
                                               _Float16* __restrict__ qh,
                                               _Float16* __restrict__ kh,
                                               _Float16* __restrict__ vt) {
  int h  = blockIdx.x & 255;
  int t  = blockIdx.x >> 8;
  int ch = t % 192;
  int b  = t / 192;
  int w  = threadIdx.x;
  const _Float16* base = qkv1 + ((size_t)b * 192 + ch) * HWP;
  const float* wdw = dw_w + ch * 9;
  float s = dw_b[ch];
#pragma unroll
  for (int dy = -1; dy <= 1; ++dy) {
    int hy = h + dy;
    if (hy < 0 || hy >= HH) continue;
#pragma unroll
    for (int dx = -1; dx <= 1; ++dx) {
      int wx = w + dx;
      if (wx < 0 || wx >= WW) continue;
      s += (float)base[hy * WW + wx] * wdw[(dy + 1) * 3 + (dx + 1)];
    }
  }
  if (ch < 128) {
    float ss = s * s;
#pragma unroll
    for (int m = 16; m >= 1; m >>= 1) ss += __shfl_xor(ss, m, 32);
    __shared__ float red[8];
    if ((threadIdx.x & 31) == 0) red[threadIdx.x >> 5] = ss;
    __syncthreads();
    float tot = 0.f;
#pragma unroll
    for (int i = 0; i < 8; ++i) tot += red[i];
    float scale = 1.0f / fmaxf(sqrtf(tot), 1e-12f);
    float val = s * scale;
    if (ch < 64) qh[((size_t)b * 64 + ch) * HWP + h * WW + w] = (_Float16)val;
    else         kh[((size_t)b * 64 + (ch - 64)) * HWP + h * WW + w] = (_Float16)val;
  } else {
    // v stored transposed [c][w][h] so it is the column-major B operand of P*V
    vt[((size_t)b * 64 + (ch - 128)) * HWP + (size_t)w * HH + h] = (_Float16)s;
  }
}

// ---------------- kernel 3: per-(b,c) attention over height ----------------
// block = 128 (4 waves); wave owns 16 rows. K/V tiles staged once per block into LDS
// via double-buffered global_load_async_to_lds_b128 and shared by all 4 waves.
__global__ __launch_bounds__(128) void attention(const _Float16* __restrict__ qh,
                                                 const _Float16* __restrict__ kh,
                                                 const _Float16* __restrict__ vt,
                                                 const float* __restrict__ temperature,
                                                 _Float16* __restrict__ ao_t) {
  __shared__ __align__(16) _Float16 pl[4 * 16 * LDK];   // per-wave P staging
  __shared__ __align__(16) _Float16 kv[2][16 * LDK];    // shared K/V tile double buffer
  int tid  = threadIdx.x;
  int iblk = blockIdx.x & 3;
  int c    = (blockIdx.x >> 2) & 63;
  int b    = blockIdx.x >> 8;
  int wave = tid >> 5;
  int lane = tid & 31;
  int hf = lane >> 4, ln = lane & 15;
  int i0 = iblk * 64 + wave * 16;
  const _Float16* qrow  = qh + ((size_t)b * 64 + c) * HWP + i0 * WW;
  const _Float16* kbase = kh + ((size_t)b * 64 + c) * HWP;
  const _Float16* vbase = vt + ((size_t)b * 64 + c) * HWP;
  float tmp = temperature[c];

  v16h qa[8];
#pragma unroll
  for (int kk = 0; kk < 8; ++kk) qa[kk] = load_a(qrow + kk * 32, WW);

  // ---- S = Q*K^T with async-staged K tiles ----
  stage_tile16(kbase, kv[0], tid);
  v8f sacc[16];
#pragma unroll
  for (int jt = 0; jt < 16; ++jt) {
    if (jt + 1 < 16) {                       // prefetch next tile into the idle buffer
      stage_tile16(kbase + (jt + 1) * 16 * WW, kv[(jt + 1) & 1], tid);
      WAIT_ASYNC_LE4();                      // in-order: current tile's 4 copies landed
    } else {
      WAIT_ASYNC_0();
    }
    __syncthreads();                         // publish tile to all waves
    const _Float16* kt = kv[jt & 1];
    v8f s = {};
#pragma unroll
    for (int kk = 0; kk < 8; ++kk)
      s = WMMA_F16(qa[kk], load_b(kt + kk * 32, LDK), s);
    sacc[jt] = s;
    __syncthreads();                         // protect buffer before next overwrite
  }

  // kick off the first two V tiles; copies overlap the softmax below
  stage_tile16(vbase,               kv[0], tid);
  stage_tile16(vbase + 16 * HH,     kv[1], tid);

  // ---- softmax per row; lanes 0-15 own rows r, lanes 16-31 own rows 8+r ----
  _Float16* myp = pl + wave * 16 * LDK;
#pragma unroll
  for (int r = 0; r < 8; ++r) {
    float mx = -3.0e38f;
#pragma unroll
    for (int jt = 0; jt < 16; ++jt) { float v = sacc[jt][r] * tmp; sacc[jt][r] = v; mx = fmaxf(mx, v); }
#pragma unroll
    for (int m = 1; m <= 8; m <<= 1) mx = fmaxf(mx, __shfl_xor(mx, m, 32));
    float sum = 0.f;
#pragma unroll
    for (int jt = 0; jt < 16; ++jt) { float e = __expf(sacc[jt][r] - mx); sacc[jt][r] = e; sum += e; }
#pragma unroll
    for (int m = 1; m <= 8; m <<= 1) sum += __shfl_xor(sum, m, 32);
    float inv = 1.0f / sum;
    int row = hf * 8 + r;
#pragma unroll
    for (int jt = 0; jt < 16; ++jt)
      myp[row * LDK + jt * 16 + ln] = (_Float16)(sacc[jt][r] * inv);
  }

  // ---- O = P*V with async-staged V tiles; one accumulator tile live at a time ----
#pragma unroll
  for (int wt = 0; wt < 16; ++wt) {
    if (wt + 1 < 16) WAIT_ASYNC_LE4(); else WAIT_ASYNC_0();
    __syncthreads();
    const _Float16* vtile = kv[wt & 1];
    v8f o = {};
#pragma unroll
    for (int kk = 0; kk < 8; ++kk)
      o = WMMA_F16(load_a(myp + kk * 32, LDK), load_b(vtile + kk * 32, LDK), o);
    int wcol = wt * 16 + ln;
#pragma unroll
    for (int r = 0; r < 8; ++r) {
      int i = i0 + hf * 8 + r;
      ao_t[(((size_t)b * HWP) + (size_t)i * WW + wcol) * 64 + c] = (_Float16)o[r];
    }
    __syncthreads();                          // all waves done reading before overwrite
    if (wt + 2 < 16) stage_tile16(vbase + (wt + 2) * 16 * HH, kv[wt & 1], tid);
  }
}

// ---------------- kernel 4: out = AO[p,64] x projW^T + b ----------------
__global__ __launch_bounds__(256) void proj(const _Float16* __restrict__ ao_t,
                                            const _Float16* __restrict__ wp,
                                            const float* __restrict__ proj_b,
                                            float* __restrict__ out) {
  int b    = blockIdx.x >> 9;
  int pblk = (blockIdx.x & 511) << 7;
  int wave = threadIdx.x >> 5;
  int lane = threadIdx.x & 31;
  int hf = lane >> 4, ln = lane & 15;
  const _Float16* abase = ao_t + ((size_t)b * HWP + pblk + wave * 16) * 64;
  v16h a0 = load_a(abase, 64);
  v16h a1 = load_a(abase + 32, 64);
#pragma unroll
  for (int ot = 0; ot < 4; ++ot) {
    float bias = proj_b[ot * 16 + ln];
    v8f acc;
#pragma unroll
    for (int r = 0; r < 8; ++r) acc[r] = bias;
    acc = WMMA_F16(a0, load_b(wp + ot * 16 * 64, 64), acc);
    acc = WMMA_F16(a1, load_b(wp + ot * 16 * 64 + 32, 64), acc);
    int o  = ot * 16 + ln;
    int p0 = pblk + wave * 16 + hf * 8;
    float* op = out + ((size_t)b * 64 + o) * HWP + p0;
#pragma unroll
    for (int r = 0; r < 8; ++r) op[r] = acc[r];   // 8 consecutive floats along p
  }
}

extern "C" void kernel_launch(void* const* d_in, const int* in_sizes, int n_in,
                              void* d_out, int out_size, void* d_ws, size_t ws_size,
                              hipStream_t stream) {
  const float* x           = (const float*)d_in[0];
  const float* qkv_w       = (const float*)d_in[1];
  const float* qkv_b       = (const float*)d_in[2];
  const float* dw_w        = (const float*)d_in[3];
  const float* dw_b        = (const float*)d_in[4];
  const float* proj_w      = (const float*)d_in[5];
  const float* proj_b      = (const float*)d_in[6];
  const float* temperature = (const float*)d_in[7];
  float* out = (float*)d_out;

  char* ws = (char*)d_ws;
  _Float16* qkv1 = (_Float16*)(ws);                     // 201,326,592 B
  _Float16* ao_t = (_Float16*)(ws);                     // alias: qkv1 dead before attention output
  _Float16* qh   = (_Float16*)(ws + 201326592);         //  67,108,864 B
  _Float16* kh   = (_Float16*)(ws + 268435456);         //  67,108,864 B
  _Float16* vt   = (_Float16*)(ws + 335544320);         //  67,108,864 B
  _Float16* wq   = (_Float16*)(ws + 402653184);         //      24,576 B
  _Float16* wp   = (_Float16*)(ws + 402677760);         //       8,192 B

  cvt_weights<<<48, 256, 0, stream>>>(qkv_w, proj_w, wq, wp);
  qkv_pointwise<<<NB * (HWP / 128), 256, 0, stream>>>(x, wq, qkv_b, qkv1);
  dw_norm<<<NB * 192 * HH, 256, 0, stream>>>(qkv1, dw_w, dw_b, qh, kh, vt);
  attention<<<NB * 64 * 4, 128, 0, stream>>>(qh, kh, vt, temperature, ao_t);
  proj<<<NB * (HWP / 128), 256, 0, stream>>>(ao_t, wp, proj_b, out);
}
